// Encoder_78915729097366
// MI455X (gfx1250) — matrix operand order repaired and checked by reading, not compiled, not run on hardware
//
#include <hip/hip_runtime.h>

typedef __attribute__((ext_vector_type(16))) _Float16 v16h;
typedef __attribute__((ext_vector_type(8)))  float    v8f;
typedef __attribute__((ext_vector_type(4)))  float    v4f;

// ---------------------------------------------------------------------------
// Weight conversion: W[k][c][d] (f32, row-major [K,Cin,Cout]) ->
// WT[d][r] (f16), r = k*Cin + c, padded to Rp (mult of 32) x Cp (mult of 16).
// W[k][c][d] = W[r*Cout + d], so this is a simple transpose+pad.
// ---------------------------------------------------------------------------
__global__ __launch_bounds__(256) void wconv_kernel(
    const float* __restrict__ W, _Float16* __restrict__ WT,
    int R, int Rp, int Cout, int Cp)
{
    int i = blockIdx.x * blockDim.x + threadIdx.x;
    int total = Rp * Cp;
    if (i >= total) return;
    int d = i / Rp;
    int r = i - d * Rp;
    float v = (r < R && d < Cout) ? W[(long)r * Cout + d] : 0.0f;
    WT[(long)d * Rp + r] = (_Float16)v;
}

// ---------------------------------------------------------------------------
// A-fragment builder for one 16-row tile at one K-step.
// A fragment (16x32 f16): lane l: M=l&15, g=l>>4; elem e -> K = 16g+e.
//   => the two 8-element runs cover K = kb+16g+[0,8) and kb+16g+8+[0,8).
//   For Cin in {8,16,32,64} each run lies inside ONE tap with contiguous
//   channels => 2 idx loads + 4 float4 gathers per K-step.
// ---------------------------------------------------------------------------
template<int CL>
__device__ __forceinline__ v16h build_a(
    const float* __restrict__ F, const int* __restrict__ idx,
    long rowBase, int mRow, int kb, int g, int R, int inRelu)
{
    constexpr int Cin = 1 << CL;
    v16h a;
    if constexpr (CL == 0) {
        // Cin==1 (conv0): scalar gathers, one tap per element
#pragma unroll
        for (int e = 0; e < 16; ++e) {
            int r  = kb + 16 * g + e;
            int rc = (r < R) ? r : (R - 1);
            int id = idx ? idx[rowBase + rc] : mRow;
            float v = F[id];
            if (inRelu) v = fmaxf(v, 0.0f);
            a[e] = (_Float16)((r < R) ? v : 0.0f);
        }
    } else {
#pragma unroll
        for (int u = 0; u < 2; ++u) {
            int  rb = kb + 16 * g + 8 * u;
            bool ok = rb < R;
            int  rc = ok ? rb : 0;
            int  k  = rc >> CL;
            int  c0 = rc & (Cin - 1);
            int  id = idx ? idx[rowBase + k] : mRow;
            const v4f* p = (const v4f*)(F + (long)id * Cin + c0);
            v4f x0 = p[0];
            v4f x1 = p[1];
#pragma unroll
            for (int t = 0; t < 4; ++t) {
                float v0 = x0[t], v1 = x1[t];
                if (inRelu) { v0 = fmaxf(v0, 0.0f); v1 = fmaxf(v1, 0.0f); }
                if (!ok)    { v0 = 0.0f; v1 = 0.0f; }
                a[8 * u + t]     = (_Float16)v0;
                a[8 * u + 4 + t] = (_Float16)v1;
            }
        }
    }
    return a;
}

// ---------------------------------------------------------------------------
// Gather-conv via WMMA f32 = f16 x f16 + f32.
// Each wave owns TWO 16-row tiles (MB=2): B fragments are loaded once and
// feed two WMMAs, halving B traffic and doubling independent WMMAs in
// flight to hide L2 gather latency. Second tile is wave-uniformly guarded,
// so EXEC is all-ones at every WMMA.
// B fragment (32x16 f16): lane l: N=l&15; elem e -> K = kb+16g+e
//   => one contiguous 32B load from WT[d][*].
// C/D: lane l: N=l&15; vgpr j -> M = 8g+j.
// ---------------------------------------------------------------------------
template<int NT, int CL>
__global__ __launch_bounds__(256) void gconv_wmma(
    const float* __restrict__ F, const int* __restrict__ idx,
    const _Float16* __restrict__ WT, const float* __restrict__ bias,
    float* __restrict__ out, const float* __restrict__ res,
    int M, int Ktaps, int R, int Rp,
    int Cout, int ldOut, int colOff, int inRelu, int outRelu)
{
    const int lane  = threadIdx.x & 31;
    const int wave  = threadIdx.x >> 5;
    const int tile0 = (blockIdx.x * 8 + wave) * 2;
    const int tiles = M >> 4;                 // M is a multiple of 16
    if (tile0 >= tiles) return;               // wave-uniform
    const bool two  = (tile0 + 1) < tiles;    // wave-uniform
    const int g     = lane >> 4;
    const int ln    = lane & 15;
    const int m0    = tile0 * 16;
    const int mRow0 = m0 + ln;
    const int mRow1 = mRow0 + 16;
    const long rowBase0 = (long)mRow0 * Ktaps;
    const long rowBase1 = (long)mRow1 * Ktaps;

    v8f acc0[NT], acc1[NT];
#pragma unroll
    for (int j = 0; j < NT; ++j) { acc0[j] = {}; acc1[j] = {}; }

    for (int kb = 0; kb < Rp; kb += 32) {
        v16h a0 = build_a<CL>(F, idx, rowBase0, mRow0, kb, g, R, inRelu);
        v16h a1;
        if (two) a1 = build_a<CL>(F, idx, rowBase1, mRow1, kb, g, R, inRelu);
#pragma unroll
        for (int j = 0; j < NT; ++j) {
            const _Float16* bp = WT + ((long)(j * 16 + ln) * Rp + kb + 16 * g);
            v16h b = *(const v16h*)bp;
            acc0[j] = __builtin_amdgcn_wmma_f32_16x16x32_f16(
                false, a0, false, b, (short)0, acc0[j], false, false);
            if (two)
                acc1[j] = __builtin_amdgcn_wmma_f32_16x16x32_f16(
                    false, a1, false, b, (short)0, acc1[j], false, false);
        }
    }

    // ---- epilogue: bias, relu, residual, column-offset concat ----
#pragma unroll
    for (int tt = 0; tt < 2; ++tt) {
        if (tt == 1 && !two) break;
        const v8f* acc = (tt == 0) ? acc0 : acc1;
        const int  mb  = m0 + tt * 16;
#pragma unroll
        for (int j = 0; j < NT; ++j) {
            int ncol = j * 16 + ln;
            if (ncol < Cout) {
                float bv = bias ? bias[ncol] : 0.0f;
#pragma unroll
                for (int jj = 0; jj < 8; ++jj) {
                    int  mr = mb + 8 * g + jj;
                    float v = acc[j][jj] + bv;
                    if (outRelu) v = fmaxf(v, 0.0f);
                    long o = (long)mr * ldOut + colOff + ncol;
                    if (res) v += res[o];
                    out[o] = v;
                }
            }
        }
    }
}

// ---------------------------------------------------------------------------
// PredictorLG mask: h = relu(f@w1+b1); logits = h@w2+b2;
// m = softmax(logits)[:,0] = 1/(1+exp(l1-l0)); out = f*(1+m).
// ---------------------------------------------------------------------------
template<int C>
__global__ __launch_bounds__(256) void mask_kernel(
    const float* __restrict__ F, float* __restrict__ out,
    const float* __restrict__ w1, const float* __restrict__ b1,
    const float* __restrict__ w2, const float* __restrict__ b2, int M)
{
    constexpr int H = C / 2;
    int m = blockIdx.x * blockDim.x + threadIdx.x;
    if (m >= M) return;
    float f[C];
#pragma unroll
    for (int i = 0; i < C; ++i) f[i] = F[(long)m * C + i];
    float l0 = b2[0], l1 = b2[1];
    for (int j = 0; j < H; ++j) {
        float hj = b1[j];
#pragma unroll 4
        for (int i = 0; i < C; ++i) hj = fmaf(f[i], w1[i * H + j], hj);
        hj = hj > 0.0f ? hj : 0.0f;
        l0 = fmaf(hj, w2[2 * j + 0], l0);
        l1 = fmaf(hj, w2[2 * j + 1], l1);
    }
    float mk = 1.0f / (1.0f + expf(l1 - l0));
    float s  = 1.0f + mk;
#pragma unroll
    for (int i = 0; i < C; ++i) out[(long)m * C + i] = f[i] * s;
}

__global__ __launch_bounds__(256) void addnoise_kernel(
    const float* __restrict__ y, const float* __restrict__ noise,
    float* __restrict__ comp, int n)
{
    int i = blockIdx.x * blockDim.x + threadIdx.x;
    if (i < n) comp[i] = y[i] + noise[i] - 0.5f;
}

// ---------------------------------------------------------------------------
// Host orchestration
// ---------------------------------------------------------------------------
namespace {
struct IrnP { const float *w00,*b00,*w01,*b01,*w10,*b10,*w11,*b11,*w12,*b12; };
struct PredP { const float *w1,*b1,*w2,*b2; };
struct WTH { const _Float16* p; int Rp; };
struct IrnH { WTH w00, w01, w10, w11, w12; };
}

extern "C" void kernel_launch(void* const* d_in, const int* in_sizes, int n_in,
                              void* d_out, int out_size, void* d_ws, size_t ws_size,
                              hipStream_t stream)
{
    (void)out_size; (void)ws_size;
    const int N0 = 150000, N1 = 60000, N2 = 24000, N3 = 10000;

    // ---- top-level input mapping (insertion-order vs fully-sorted flatten) ----
    const float *x_feat, *noise;
    const int *nbr0, *dn0, *nbr1, *dn1, *nbr2, *dn2, *nbr3;
    int pbase;
    bool topInsertion = (n_in > 1 && in_sizes[0] == N0 && in_sizes[1] == N3 * 8);
    if (topInsertion) {
        x_feat = (const float*)d_in[0]; noise = (const float*)d_in[1];
        nbr0 = (const int*)d_in[2]; dn0 = (const int*)d_in[3];
        nbr1 = (const int*)d_in[4]; dn1 = (const int*)d_in[5];
        nbr2 = (const int*)d_in[6]; dn2 = (const int*)d_in[7];
        nbr3 = (const int*)d_in[8]; pbase = 9;
    } else {
        dn0 = (const int*)d_in[0]; dn1 = (const int*)d_in[1]; dn2 = (const int*)d_in[2];
        nbr0 = (const int*)d_in[3]; nbr1 = (const int*)d_in[4];
        nbr2 = (const int*)d_in[5]; nbr3 = (const int*)d_in[6];
        noise = (const float*)d_in[7]; pbase = 8;
        x_feat = (const float*)d_in[n_in - 1];
    }

    // ---- parameter leaf mapping ----
    const float *conv0_w,*conv0_b,*conv1_w,*conv1_b,*conv2_w,*conv2_b,*conv3_w,*conv3_b;
    const float *down0_w,*down0_b,*down1_w,*down1_b,*down2_w,*down2_b;
    IrnP blk[3][3]; PredP pr[4];
    bool sortedParams = (in_sizes[pbase] <= 32);  // b00 (8/16) vs conv0_w (432)
    int q = pbase;
    if (sortedParams) { // jax tree_flatten: dict keys sorted, DFS
        for (int B = 0; B < 3; ++B)
            for (int i = 0; i < 3; ++i) {
                IrnP& p = blk[B][i];
                p.b00=(const float*)d_in[q++]; p.b01=(const float*)d_in[q++];
                p.b10=(const float*)d_in[q++]; p.b11=(const float*)d_in[q++];
                p.b12=(const float*)d_in[q++];
                p.w00=(const float*)d_in[q++]; p.w01=(const float*)d_in[q++];
                p.w10=(const float*)d_in[q++]; p.w11=(const float*)d_in[q++];
                p.w12=(const float*)d_in[q++];
            }
        conv0_b=(const float*)d_in[q++]; conv0_w=(const float*)d_in[q++];
        conv1_b=(const float*)d_in[q++]; conv1_w=(const float*)d_in[q++];
        conv2_b=(const float*)d_in[q++]; conv2_w=(const float*)d_in[q++];
        conv3_b=(const float*)d_in[q++]; conv3_w=(const float*)d_in[q++];
        down0_b=(const float*)d_in[q++]; down0_w=(const float*)d_in[q++];
        down1_b=(const float*)d_in[q++]; down1_w=(const float*)d_in[q++];
        down2_b=(const float*)d_in[q++]; down2_w=(const float*)d_in[q++];
        for (int p = 0; p < 4; ++p) {
            pr[p].b1=(const float*)d_in[q++]; pr[p].b2=(const float*)d_in[q++];
            pr[p].w1=(const float*)d_in[q++]; pr[p].w2=(const float*)d_in[q++];
        }
    } else { // dict insertion order
        auto irnf = [&](IrnP& p) {
            p.w00=(const float*)d_in[q++]; p.b00=(const float*)d_in[q++];
            p.w01=(const float*)d_in[q++]; p.b01=(const float*)d_in[q++];
            p.w10=(const float*)d_in[q++]; p.b10=(const float*)d_in[q++];
            p.w11=(const float*)d_in[q++]; p.b11=(const float*)d_in[q++];
            p.w12=(const float*)d_in[q++]; p.b12=(const float*)d_in[q++];
        };
        auto prf = [&](PredP& p) {
            p.w1=(const float*)d_in[q++]; p.b1=(const float*)d_in[q++];
            p.w2=(const float*)d_in[q++]; p.b2=(const float*)d_in[q++];
        };
        conv0_w=(const float*)d_in[q++]; conv0_b=(const float*)d_in[q++];
        prf(pr[0]);
        down0_w=(const float*)d_in[q++]; down0_b=(const float*)d_in[q++];
        for (int i = 0; i < 3; ++i) irnf(blk[0][i]);
        prf(pr[1]);
        conv1_w=(const float*)d_in[q++]; conv1_b=(const float*)d_in[q++];
        down1_w=(const float*)d_in[q++]; down1_b=(const float*)d_in[q++];
        for (int i = 0; i < 3; ++i) irnf(blk[1][i]);
        prf(pr[2]);
        conv2_w=(const float*)d_in[q++]; conv2_b=(const float*)d_in[q++];
        down2_w=(const float*)d_in[q++]; down2_b=(const float*)d_in[q++];
        for (int i = 0; i < 3; ++i) irnf(blk[2][i]);
        prf(pr[3]);
        conv3_w=(const float*)d_in[q++]; conv3_b=(const float*)d_in[q++];
    }

    // ---- workspace layout (all 16B-aligned for float4 gathers) ----
    char* ws = (char*)d_ws;
    _Float16* W16 = (_Float16*)ws;                     // <=1MB of f16 weights
    float* FB0 = (float*)(ws + (1 << 20));             // N0*16
    float* FB1 = FB0 + 2400000;                        // max(N1*32, N2*64)
    float* FB2 = FB1 + 1920000;
    float* T0  = FB2 + 1920000;                        // max 480000 floats
    float* T1  = T0  + 500000;

    // ---- output slices: (y, out1, out0, compressed) concatenated ----
    float* outF = (float*)d_out;
    float* y_ptr    = outF;
    float* out1_ptr = outF + (long)N3 * 8;
    float* out0_ptr = out1_ptr + (long)N2 * 64;
    float* comp_ptr = out0_ptr + (long)N1 * 32;

    // ---- weight conversion (f32 -> padded transposed f16) ----
    long woff = 0;
    auto cvt = [&](const float* W, int Ktaps, int Cin, int Cout) -> WTH {
        int R = Ktaps * Cin, Rp = (R + 31) & ~31, Cp = (Cout + 15) & ~15;
        _Float16* dst = W16 + woff;
        int total = Rp * Cp;
        wconv_kernel<<<dim3((total + 255) / 256), dim3(256), 0, stream>>>(
            W, dst, R, Rp, Cout, Cp);
        woff += total;
        WTH h; h.p = dst; h.Rp = Rp; return h;
    };
    WTH hconv0 = cvt(conv0_w, 27, 1, 16);
    WTH hdown0 = cvt(down0_w, 8, 16, 32);
    WTH hconv1 = cvt(conv1_w, 27, 32, 32);
    WTH hdown1 = cvt(down1_w, 8, 32, 64);
    WTH hconv2 = cvt(conv2_w, 27, 64, 64);
    WTH hdown2 = cvt(down2_w, 8, 64, 32);
    WTH hconv3 = cvt(conv3_w, 27, 32, 8);
    IrnH bh[3][3];
    const int cs[3] = {32, 64, 32};
    for (int B = 0; B < 3; ++B) {
        int c = cs[B], c4 = c / 4, c2 = c / 2;
        for (int i = 0; i < 3; ++i) {
            bh[B][i].w00 = cvt(blk[B][i].w00, 27, c,  c4);
            bh[B][i].w01 = cvt(blk[B][i].w01, 27, c4, c2);
            bh[B][i].w10 = cvt(blk[B][i].w10, 1,  c,  c4);
            bh[B][i].w11 = cvt(blk[B][i].w11, 27, c4, c4);
            bh[B][i].w12 = cvt(blk[B][i].w12, 1,  c4, c2);
        }
    }

    // ---- gather-conv dispatch over (NT, log2 Cin); MB=2 tiles per wave ----
    auto gconv = [&](const float* F, const int* idx, WTH h, const float* bias,
                     float* out, const float* res, int M, int Cin, int Ktaps,
                     int Cout, int ldOut, int colOff, int inRelu, int outRelu) {
        int R = Ktaps * Cin;
        int cl = 31 - __builtin_clz((unsigned)Cin);
        int tiles = M / 16;
        int grid = (tiles + 15) / 16;          // 8 waves * 2 tiles per block
        int nT = (Cout + 15) / 16;
#define GC_LAUNCH(NT_, CL_)                                                    \
        gconv_wmma<NT_, CL_><<<dim3(grid), dim3(256), 0, stream>>>(            \
            F, idx, h.p, bias, out, res, M, Ktaps, R, h.Rp, Cout, ldOut,       \
            colOff, inRelu, outRelu)
        int key = nT * 16 + cl;
        switch (key) {
            case 16 + 0: GC_LAUNCH(1, 0); break;
            case 16 + 3: GC_LAUNCH(1, 3); break;
            case 16 + 4: GC_LAUNCH(1, 4); break;
            case 16 + 5: GC_LAUNCH(1, 5); break;
            case 16 + 6: GC_LAUNCH(1, 6); break;
            case 32 + 3: GC_LAUNCH(2, 3); break;
            case 32 + 4: GC_LAUNCH(2, 4); break;
            case 32 + 5: GC_LAUNCH(2, 5); break;
            case 32 + 6: GC_LAUNCH(2, 6); break;
            case 64 + 3: GC_LAUNCH(4, 3); break;
            case 64 + 4: GC_LAUNCH(4, 4); break;
            case 64 + 5: GC_LAUNCH(4, 5); break;
            case 64 + 6: GC_LAUNCH(4, 6); break;
            default: break;
        }
#undef GC_LAUNCH
    };
    auto maskl = [&](const float* F, float* out, const PredP& p, int c, int M) {
        int grid = (M + 255) / 256;
        if (c == 16)
            mask_kernel<16><<<dim3(grid), dim3(256), 0, stream>>>(F, out, p.w1, p.b1, p.w2, p.b2, M);
        else if (c == 32)
            mask_kernel<32><<<dim3(grid), dim3(256), 0, stream>>>(F, out, p.w1, p.b1, p.w2, p.b2, M);
        else
            mask_kernel<64><<<dim3(grid), dim3(256), 0, stream>>>(F, out, p.w1, p.b1, p.w2, p.b2, M);
    };
    auto irn = [&](const float* Fin, float* Fout, const IrnP& p, const IrnH& h,
                   const int* idx, int M, int c) {
        int c4 = c / 4, c2 = c / 2;
        // branch a: c -> c4 (relu) -> c2, into Fout[:, c2:c] + residual
        gconv(Fin, idx,     h.w00, p.b00, T0,   nullptr, M, c,  27, c4, c4, 0,  0, 1);
        gconv(T0,  idx,     h.w01, p.b01, Fout, Fin,     M, c4, 27, c2, c,  c2, 0, 0);
        // branch b: lin c->c4 relu, conv c4->c4 relu, lin c4->c2, into Fout[:, 0:c2] + residual
        gconv(Fin, nullptr, h.w10, p.b10, T1,   nullptr, M, c,  1,  c4, c4, 0,  0, 1);
        gconv(T1,  idx,     h.w11, p.b11, T0,   nullptr, M, c4, 27, c4, c4, 0,  0, 1);
        gconv(T0,  nullptr, h.w12, p.b12, Fout, Fin,     M, c4, 1,  c2, c,  0,  0, 0);
    };

    // ---- forward pass ----
    gconv(x_feat, nbr0, hconv0, conv0_b, FB0, nullptr, N0, 1, 27, 16, 16, 0, 0, 0);
    maskl(FB0, FB0, pr[0], 16, N0);                                  // f = f*(1+m)
    gconv(FB0, dn0, hdown0, down0_b, FB1, nullptr, N1, 16, 8, 32, 32, 0, 1, 1);

    float* cur = FB1; float* nxt = FB2;
    for (int i = 0; i < 3; ++i) { irn(cur, nxt, blk[0][i], bh[0][i], nbr1, N1, 32);
                                  float* t = cur; cur = nxt; nxt = t; }       // -> FB2
    maskl(cur, out0_ptr, pr[1], 32, N1);                             // out0

    gconv(out0_ptr, nbr1, hconv1, conv1_b, FB1, nullptr, N1, 32, 27, 32, 32, 0, 0, 0);
    gconv(FB1, dn1, hdown1, down1_b, FB2, nullptr, N2, 32, 8, 64, 64, 0, 1, 1);

    cur = FB2; nxt = FB1;
    for (int i = 0; i < 3; ++i) { irn(cur, nxt, blk[1][i], bh[1][i], nbr2, N2, 64);
                                  float* t = cur; cur = nxt; nxt = t; }       // -> FB1
    maskl(cur, out1_ptr, pr[2], 64, N2);                             // out1

    gconv(out1_ptr, nbr2, hconv2, conv2_b, FB1, nullptr, N2, 64, 27, 64, 64, 0, 0, 0);
    gconv(FB1, dn2, hdown2, down2_b, FB2, nullptr, N3, 64, 8, 32, 32, 0, 1, 1);

    cur = FB2; nxt = FB1;
    for (int i = 0; i < 3; ++i) { irn(cur, nxt, blk[2][i], bh[2][i], nbr3, N3, 32);
                                  float* t = cur; cur = nxt; nxt = t; }       // -> FB1
    maskl(cur, T1, pr[3], 32, N3);                                   // out2

    gconv(T1, nbr3, hconv3, conv3_b, y_ptr, nullptr, N3, 32, 27, 8, 8, 0, 0, 0);  // y
    addnoise_kernel<<<dim3((N3 * 8 + 255) / 256), dim3(256), 0, stream>>>(
        y_ptr, noise, comp_ptr, N3 * 8);                             // compressed
}
